// GRU_CLF_50689204028006
// MI455X (gfx1250) — matrix operand order repaired and checked by reading, not compiled, not run on hardware
//
#include <hip/hip_runtime.h>
#include <hip/hip_bf16.h>
#include <stdint.h>

// ---------------------------------------------------------------------------
// Types for CDNA5 WMMA (wave32): V_WMMA_F32_16X16X32_BF16
// ---------------------------------------------------------------------------
typedef __bf16 bf16_t;
typedef __attribute__((ext_vector_type(16))) __bf16 v16bf;
typedef __attribute__((ext_vector_type(8)))  float  v8f;

#define T_STEPS 512
#define BATCH   1024
#define HID     64
#define G3H     192   // 3*H, gate order r,z,n

union Frag32 { v16bf v; uint4 q[2]; };

// A-matrix fragment (16x32 bf16, MxK): lane L holds row M=L%16,
// K = [8*(L/16), +8) and [16+8*(L/16), +8)  -> two 16B chunks.
__device__ __forceinline__ v16bf load_a_frag(const bf16_t* p, int m, int kb, int rs) {
  Frag32 f;
  f.q[0] = *(const uint4*)(p + (size_t)m * rs + kb);
  f.q[1] = *(const uint4*)(p + (size_t)m * rs + kb + 16);
  return f.v;
}

// B-matrix fragment (32x16 bf16, KxN) for B[k][n] = W[n][k] with W row-major
// [192, 64]: lane L holds col N=L%16, K = [kb, kb+8) and [kb+16, kb+24).
__device__ __forceinline__ v16bf load_b_frag(const bf16_t* W, int n, int kb) {
  Frag32 f;
  const bf16_t* q = W + (size_t)n * HID + kb;
  f.q[0] = *(const uint4*)(q);
  f.q[1] = *(const uint4*)(q + 16);
  return f.v;
}

__device__ __forceinline__ v8f wmma_bf16(v16bf a, v16bf b, v8f c) {
  // emits v_wmma_f32_16x16x32_bf16
  return __builtin_amdgcn_wmma_f32_16x16x32_bf16(false, a, false, b, (short)0, c,
                                                 false, false);
}

__device__ __forceinline__ float fast_sig(float x) {
  // 1/(1+exp(-x)) via v_exp_f32 + v_rcp_f32
  return __builtin_amdgcn_rcpf(1.0f + __builtin_amdgcn_exp2f(x * -1.4426950408889634f));
}
__device__ __forceinline__ float fast_tanh(float x) {
  return 2.0f * fast_sig(2.0f * x) - 1.0f;
}

// Split barrier halves (ISA 01_flow_sync §6). Signal carries an explicit
// dscnt wait so this wave's LDS stores are visible before others pass wait.
__device__ __forceinline__ void barrier_signal_lds() {
  asm volatile("s_wait_dscnt 0x0\n\ts_barrier_signal -1" ::: "memory");
}
__device__ __forceinline__ void barrier_wait() {
  asm volatile("s_barrier_wait -1" ::: "memory");
}

// 3 gate-tiles x 2 K-chunks, C starts at inline 0
__device__ __forceinline__ void gemm3(v8f out[3], v16bf a0, v16bf a1,
                                      const v16bf w[3][2]) {
  #pragma unroll
  for (int g = 0; g < 3; ++g) {
    v8f c = {};
    c = wmma_bf16(a0, w[g][0], c);
    c = wmma_bf16(a1, w[g][1], c);
    out[g] = c;
  }
}

// GRU gate epilogue (lane-local; r/z biases pre-summed, n-gate biases split)
__device__ __forceinline__ v8f gru_gate(const v8f gi[3], const v8f gh[3],
                                        float brz0, float brz1, float bi2,
                                        float bh2, v8f hold) {
  v8f hnew;
  #pragma unroll
  for (int i = 0; i < 8; ++i) {
    float r = fast_sig(gi[0][i] + gh[0][i] + brz0);
    float z = fast_sig(gi[1][i] + gh[1][i] + brz1);
    float n = fast_tanh(gi[2][i] + bi2 + r * (gh[2][i] + bh2));
    hnew[i] = n + z * (hold[i] - n);     // == (1-z)*n + z*h
  }
  return hnew;
}

__device__ __forceinline__ void store_h(bf16_t* buf, int lh, int col, v8f h) {
  #pragma unroll
  for (int i = 0; i < 8; ++i)              // C/D layout: VGPR i -> M = i+8*half
    buf[(i + 8 * lh) * HID + col] = (bf16_t)h[i];
}

__device__ __forceinline__ void load_w3(v16bf w[3][2], const bf16_t* W,
                                        int w4, int lm, int lh) {
  #pragma unroll
  for (int g = 0; g < 3; ++g) {
    const int n = (w4 + 4 * g) * 16 + lm;  // N-tiles {w, w+4, w+8}
    #pragma unroll
    for (int kc = 0; kc < 2; ++kc)
      w[g][kc] = load_b_frag(W, n, kc * 32 + lh * 8);
  }
}

// ---------------------------------------------------------------------------
// Kernel 1: f32 -> bf16 weight conversion
// ---------------------------------------------------------------------------
__global__ void f32_to_bf16_kernel(const float* __restrict__ in,
                                   bf16_t* __restrict__ out, int n) {
  int i = blockIdx.x * 256 + threadIdx.x;
  if (i < n) out[i] = (bf16_t)in[i];
}

// ---------------------------------------------------------------------------
// Kernel 2: x [B, 64, T] f32 -> xt [T, B, 64] bf16 (LDS tile transpose)
// ---------------------------------------------------------------------------
__global__ __launch_bounds__(256) void transpose_x_kernel(
    const float* __restrict__ x, bf16_t* __restrict__ xt) {
  const int b  = blockIdx.x;        // 0..1023
  const int t0 = blockIdx.y * 64;   // 8 chunks of 64 along T
  __shared__ bf16_t tile[64][66];   // pad to dodge bank conflicts
  #pragma unroll
  for (int it = 0; it < 16; ++it) {
    int flat = it * 256 + threadIdx.x;    // 0..4095
    int c = flat >> 6, tt = flat & 63;    // coalesced along T
    float v = x[((size_t)b * 64 + c) * T_STEPS + t0 + tt];
    tile[c][tt] = (bf16_t)v;
  }
  __syncthreads();
  #pragma unroll
  for (int it = 0; it < 16; ++it) {
    int flat = it * 256 + threadIdx.x;
    int tt = flat >> 6, c = flat & 63;    // coalesced along C
    xt[((size_t)(t0 + tt) * BATCH + b) * HID + c] = tile[c][tt];
  }
}

// ---------------------------------------------------------------------------
// Kernel 3: BOTH GRU layers, fused + wave-specialized.
// Block = 16 batch rows, 256 threads (8 waves): waves 0-3 run layer0 step t,
// waves 4-7 run layer1 step t-1 concurrently (2 waves/SIMD). `wave` is
// readfirstlane'd so the layer split is a *scalar* branch (no EXEC masking).
// Split barrier: signal after LDS h store, global x prefetch issued between
// signal and wait, wait right before next step's LDS reads.
// ---------------------------------------------------------------------------
__global__ __launch_bounds__(256) void gru2_fused_kernel(
    const bf16_t* __restrict__ xseq,   // [T, B, 64] bf16
    const bf16_t* __restrict__ Wih0, const bf16_t* __restrict__ Whh0,
    const float*  __restrict__ bih0, const float*  __restrict__ bhh0,
    const bf16_t* __restrict__ Wih1, const bf16_t* __restrict__ Whh1,
    const float*  __restrict__ bih1, const float*  __restrict__ bhh1,
    float*        __restrict__ hlast)  // [B, 64] f32
{
  const int lane  = threadIdx.x & 31;
  // wave index is uniform per wave: readfirstlane makes that explicit -> SGPR
  const int wave  = __builtin_amdgcn_readfirstlane(threadIdx.x) >> 5;  // 0..7
  const int w4    = wave & 3;           // N-tile group within the layer
  const bool L0   = wave < 4;           // layer assignment (scalar branch)
  const int bbase = blockIdx.x * 16;    // batch tile
  const int lm    = lane & 15;          // row (A) / col (B,C,D) index in tile
  const int lh    = lane >> 4;          // lane half
  const int col   = w4 * 16 + lm;       // owned h column

  __shared__ __align__(16) bf16_t h0sm[2][16 * HID];  // layer0 h, double-buf
  __shared__ __align__(16) bf16_t h1sm[2][16 * HID];  // layer1 h, double-buf

  // Per-wave weights/biases: only this wave's layer
  const bf16_t* Wi  = L0 ? Wih0 : Wih1;
  const bf16_t* Wh  = L0 ? Whh0 : Whh1;
  const float*  bip = L0 ? bih0 : bih1;
  const float*  bhp = L0 ? bhh0 : bhh1;
  v16bf wi[3][2], wh[3][2];
  load_w3(wi, Wi, w4, lm, lh);
  load_w3(wh, Wh, w4, lm, lh);
  float bi[3], bh[3];
  #pragma unroll
  for (int g = 0; g < 3; ++g) { bi[g] = bip[col + 64 * g]; bh[g] = bhp[col + 64 * g]; }
  const float brz0 = bi[0] + bh[0], brz1 = bi[1] + bh[1];

  // h0(0) read from h0sm[0] at t=0 ; h1(0) state read from h1sm[1] at t=1
  for (int i = threadIdx.x; i < 16 * HID; i += 256) {
    h0sm[0][i] = (bf16_t)0.0f;
    h1sm[1][i] = (bf16_t)0.0f;
  }
  v8f hreg = {};                         // this wave's owned h entries (f32)

  // layer0 waves prefetch x fragments for t=0
  v16bf ax0 = {}, ax1 = {};
  if (L0) {
    const bf16_t* xp = xseq + (size_t)bbase * HID;
    ax0 = load_a_frag(xp, lm, 0  + lh * 8, HID);
    ax1 = load_a_frag(xp, lm, 32 + lh * 8, HID);
  }
  __syncthreads();

  // ---- prologue t=0: layer0 waves only ----
  if (L0) {
    const bf16_t* xp = xseq + ((size_t)BATCH + bbase) * HID;   // t=1
    v16bf nx0 = load_a_frag(xp, lm, 0  + lh * 8, HID);
    v16bf nx1 = load_a_frag(xp, lm, 32 + lh * 8, HID);
    v16bf a0 = load_a_frag(h0sm[0], lm, 0  + lh * 8, HID);
    v16bf a1 = load_a_frag(h0sm[0], lm, 32 + lh * 8, HID);
    v8f gi[3], gh[3];
    gemm3(gi, ax0, ax1, wi);
    gemm3(gh, a0, a1, wh);
    hreg = gru_gate(gi, gh, brz0, brz1, bi[2], bh[2], hreg);
    store_h(h0sm[1], lh, col, hreg);
    ax0 = nx0; ax1 = nx1;
  }
  __syncthreads();

  // ---- steady state: iteration t = layer0(t) || layer1(t-1) ----
  for (int t = 1; t < T_STEPS; ++t) {
    const int p = t & 1;
    if (L0) {
      v16bf a0 = load_a_frag(h0sm[p], lm, 0  + lh * 8, HID);
      v16bf a1 = load_a_frag(h0sm[p], lm, 32 + lh * 8, HID);
      v8f gi[3], gh[3];
      gemm3(gi, ax0, ax1, wi);    // input projection (prefetched x)
      gemm3(gh, a0, a1, wh);      // recurrent projection (critical path)
      hreg = gru_gate(gi, gh, brz0, brz1, bi[2], bh[2], hreg);
      store_h(h0sm[p ^ 1], lh, col, hreg);
    } else {
      // layer1 step t-1: input = h0(t-1) (in h0sm[p]), state = h1(t-2)
      v16bf a00 = load_a_frag(h0sm[p], lm, 0  + lh * 8, HID);
      v16bf a01 = load_a_frag(h0sm[p], lm, 32 + lh * 8, HID);
      v16bf a10 = load_a_frag(h1sm[p], lm, 0  + lh * 8, HID);
      v16bf a11 = load_a_frag(h1sm[p], lm, 32 + lh * 8, HID);
      v8f gi[3], gh[3];
      gemm3(gi, a00, a01, wi);
      gemm3(gh, a10, a11, wh);
      hreg = gru_gate(gi, gh, brz0, brz1, bi[2], bh[2], hreg);
      store_h(h1sm[p ^ 1], lh, col, hreg);
    }

    barrier_signal_lds();         // LDS stores visible, arrive at barrier

    if (L0) {                     // overlap with barrier: prefetch x(t+1)
      const int tn = (t + 1 < T_STEPS) ? t + 1 : t;
      const bf16_t* xp = xseq + ((size_t)tn * BATCH + bbase) * HID;
      ax0 = load_a_frag(xp, lm, 0  + lh * 8, HID);
      ax1 = load_a_frag(xp, lm, 32 + lh * 8, HID);
    }

    barrier_wait();               // release: next step's LDS reads may begin
  }

  // ---- epilogue: layer1 waves compute step T-1 and write h_last ----
  if (!L0) {
    const int p = T_STEPS & 1;    // == 0; buffers written at t = T-1
    v16bf a00 = load_a_frag(h0sm[p], lm, 0  + lh * 8, HID);
    v16bf a01 = load_a_frag(h0sm[p], lm, 32 + lh * 8, HID);
    v16bf a10 = load_a_frag(h1sm[p], lm, 0  + lh * 8, HID);
    v16bf a11 = load_a_frag(h1sm[p], lm, 32 + lh * 8, HID);
    v8f gi[3], gh[3];
    gemm3(gi, a00, a01, wi);
    gemm3(gh, a10, a11, wh);
    v8f hf = gru_gate(gi, gh, brz0, brz1, bi[2], bh[2], hreg);
    #pragma unroll
    for (int i = 0; i < 8; ++i)
      hlast[(size_t)(bbase + i + 8 * lh) * HID + col] = hf[i];
  }
}

// ---------------------------------------------------------------------------
// Kernel 4: y = relu(h_last @ W1^T + b1) @ W2^T + b2   (tiny, VALU)
// ---------------------------------------------------------------------------
__global__ __launch_bounds__(256) void mlp_kernel(
    const float* __restrict__ hlast, const float* __restrict__ W1,
    const float* __restrict__ b1, const float* __restrict__ W2,
    const float* __restrict__ b2, float* __restrict__ out) {
  int b = blockIdx.x * 256 + threadIdx.x;
  if (b >= BATCH) return;
  float h[HID];
  #pragma unroll
  for (int i = 0; i < HID; ++i) h[i] = hlast[(size_t)b * HID + i];
  float acc[4] = {b2[0], b2[1], b2[2], b2[3]};
  for (int j = 0; j < 32; ++j) {
    float s = b1[j];
    #pragma unroll
    for (int i = 0; i < HID; ++i) s += h[i] * W1[j * HID + i];
    s = fmaxf(s, 0.0f);
    #pragma unroll
    for (int k = 0; k < 4; ++k) acc[k] += s * W2[k * 32 + j];
  }
  #pragma unroll
  for (int k = 0; k < 4; ++k) out[b * 4 + k] = acc[k];
}

// ---------------------------------------------------------------------------
// Host launcher
// ---------------------------------------------------------------------------
extern "C" void kernel_launch(void* const* d_in, const int* in_sizes, int n_in,
                              void* d_out, int out_size, void* d_ws, size_t ws_size,
                              hipStream_t stream) {
  (void)in_sizes; (void)n_in; (void)out_size; (void)ws_size;

  const float* x     = (const float*)d_in[0];
  const float* W_ih0 = (const float*)d_in[1];
  const float* W_hh0 = (const float*)d_in[2];
  const float* b_ih0 = (const float*)d_in[3];
  const float* b_hh0 = (const float*)d_in[4];
  const float* W_ih1 = (const float*)d_in[5];
  const float* W_hh1 = (const float*)d_in[6];
  const float* b_ih1 = (const float*)d_in[7];
  const float* b_hh1 = (const float*)d_in[8];
  const float* W1    = (const float*)d_in[9];
  const float* b1    = (const float*)d_in[10];
  const float* W2    = (const float*)d_in[11];
  const float* b2    = (const float*)d_in[12];

  // Workspace layout (256B aligned): xt (64MB) | bf16 weights | hlast
  char* ws = (char*)d_ws;
  const size_t SEQ_BYTES = (size_t)T_STEPS * BATCH * HID * sizeof(bf16_t); // 64 MB
  const size_t W_ELEMS   = (size_t)G3H * HID;                              // 12288
  bf16_t* xt    = (bf16_t*)(ws);
  bf16_t* wih0b = (bf16_t*)(ws + SEQ_BYTES);
  bf16_t* whh0b = wih0b + W_ELEMS;
  bf16_t* wih1b = whh0b + W_ELEMS;
  bf16_t* whh1b = wih1b + W_ELEMS;
  float*  hlast = (float*)(ws + SEQ_BYTES + 4 * W_ELEMS * sizeof(bf16_t));

  // 1) weights -> bf16
  const int nconv = (int)W_ELEMS, cblk = (nconv + 255) / 256;
  f32_to_bf16_kernel<<<cblk, 256, 0, stream>>>(W_ih0, wih0b, nconv);
  f32_to_bf16_kernel<<<cblk, 256, 0, stream>>>(W_hh0, whh0b, nconv);
  f32_to_bf16_kernel<<<cblk, 256, 0, stream>>>(W_ih1, wih1b, nconv);
  f32_to_bf16_kernel<<<cblk, 256, 0, stream>>>(W_hh1, whh1b, nconv);

  // 2) x -> [T, B, 64] bf16
  dim3 tg(BATCH, T_STEPS / 64);
  transpose_x_kernel<<<tg, 256, 0, stream>>>(x, xt);

  // 3) fused, wave-specialized 2-layer GRU recurrence (single pass)
  gru2_fused_kernel<<<BATCH / 16, 256, 0, stream>>>(
      xt, wih0b, whh0b, b_ih0, b_hh0, wih1b, whh1b, b_ih1, b_hh1, hlast);

  // 4) classifier head
  mlp_kernel<<<BATCH / 256, 256, 0, stream>>>(hlast, W1, b1, W2, b2,
                                              (float*)d_out);
}
